// ANEEAttentionLayer_33466385170719
// MI455X (gfx1250) — matrix operand order repaired and checked by reading, not compiled, not run on hardware
//
#include <hip/hip_runtime.h>

#define N_NODES 10000
#define N_EDGES 640000
#define DIM 128
#define ALPHA 0.3f

typedef __attribute__((ext_vector_type(16))) _Float16 v16h;
typedef __attribute__((ext_vector_type(8)))  _Float16 v8h;
typedef __attribute__((ext_vector_type(8)))  float    v8f;
typedef __attribute__((ext_vector_type(4)))  float    v4f;

__device__ __forceinline__ v8f wmma_f16(v16h a, v16h b, v8f c) {
  // 8 args: (neg_a, A, neg_b, B, c_mod, C, reuse_a, reuse_b)
  return __builtin_amdgcn_wmma_f32_16x16x32_f16(false, a, false, b, (short)0, c,
                                                false, false);
}

// Opaque zero: defeats LICM so loop-invariant LDS fragment loads stay in the
// loop (ds_load_b128) instead of being hoisted into ~256 VGPRs and spilled to
// scratch memory.
__device__ __forceinline__ int opaque_zero() {
  int z;
  asm volatile("v_mov_b32 %0, 0" : "=v"(z));
  return z;
}

// load 8 consecutive f32 from global, convert to 8 halves
__device__ __forceinline__ v8h cvt8(const float* p) {
  v4f a = *(const v4f*)p;
  v4f b = *(const v4f*)(p + 4);
  v8h r;
#pragma unroll
  for (int i = 0; i < 4; ++i) { r[i] = (_Float16)a[i]; r[4 + i] = (_Float16)b[i]; }
  return r;
}

__device__ __forceinline__ v16h cat16(v8h lo, v8h hi) {
  v16h r;
#pragma unroll
  for (int i = 0; i < 8; ++i) { r[i] = lo[i]; r[8 + i] = hi[i]; }
  return r;
}

__device__ __forceinline__ v8h lds8(const _Float16* p) { return *(const v8h*)p; }

// ---------------------------------------------------------------------------
// Kernel 1: h = leaky(nodeF @ Wu + b); s1 = h . a_w[0:128]; s2 = h . a_w[128:256]
// ---------------------------------------------------------------------------
__global__ void node_score_kernel(const float* __restrict__ nodeF,
                                  const float* __restrict__ Wu,
                                  const float* __restrict__ Wub,
                                  const float* __restrict__ aw,
                                  float* __restrict__ s1,
                                  float* __restrict__ s2) {
  extern __shared__ char smem[];
  _Float16* sW = (_Float16*)smem;              // DIM*DIM halves, B-frag layout: [n*DIM + k]
  float* sA1 = (float*)(sW + DIM * DIM);
  float* sA2 = sA1 + DIM;
  float* sB  = sA2 + DIM;

  const int tid = threadIdx.x;
  // stage Wu as f16, column-major (K contiguous per output column n)
  for (int t = tid; t < 2048; t += 256) {
    int n = t >> 4, k0 = (t & 15) * 8;
    v8h p;
#pragma unroll
    for (int j = 0; j < 8; ++j) p[j] = (_Float16)Wu[(k0 + j) * DIM + n];
    *(v8h*)(sW + n * DIM + k0) = p;
  }
  for (int t = tid; t < DIM; t += 256) {
    sA1[t] = aw[t];
    sA2[t] = aw[DIM + t];
    sB[t]  = Wub[t];
  }
  __syncthreads();

  const int lane = tid & 31;
  const int col  = lane & 15;
  const int half = lane >> 4;
  const int wave = tid >> 5;
  const int nwaves = gridDim.x * 8;

  for (int tile = blockIdx.x * 8 + wave; tile < N_NODES / 16; tile += nwaves) {
    int zf = opaque_zero();   // per-iteration opaque 0: keep LDS reads in-loop
    int row = tile * 16 + col;
    const float* rp = nodeF + (long)row * DIM;
    // A fragments (16x32 f16 tiles): e<8 -> K=32kt+8h+e ; e>=8 -> K=32kt+16+8h+(e-8)
    v16h afrag[4];
#pragma unroll
    for (int kt = 0; kt < 4; ++kt)
      afrag[kt] = cat16(cvt8(rp + 32 * kt + 8 * half),
                        cvt8(rp + 32 * kt + 16 + 8 * half));

    float p1[8], p2[8];
#pragma unroll
    for (int r = 0; r < 8; ++r) { p1[r] = 0.f; p2[r] = 0.f; }

#pragma unroll
    for (int nt = 0; nt < 8; ++nt) {
      v8f acc = {};
      const _Float16* bp = sW + (16 * nt + col) * DIM + 8 * half + zf;
#pragma unroll
      for (int kt = 0; kt < 4; ++kt) {
        v16h bfrag = cat16(lds8(bp + 32 * kt), lds8(bp + 32 * kt + 16));
        acc = wmma_f16(afrag[kt], bfrag, acc);
      }
      float a1 = sA1[16 * nt + col];
      float a2 = sA2[16 * nt + col];
      float bb = sB[16 * nt + col];
#pragma unroll
      for (int r = 0; r < 8; ++r) {
        float hv = acc[r] + bb;
        hv = hv >= 0.f ? hv : ALPHA * hv;
        p1[r] += hv * a1;
        p2[r] += hv * a2;
      }
    }
    // reduce over the 16 lanes holding columns of each row
#pragma unroll
    for (int r = 0; r < 8; ++r) {
#pragma unroll
      for (int m = 1; m < 16; m <<= 1) {
        p1[r] += __shfl_xor(p1[r], m, 32);
        p2[r] += __shfl_xor(p2[r], m, 32);
      }
    }
    if (col == 0) {
#pragma unroll
      for (int r = 0; r < 8; ++r) {
        int node = tile * 16 + 8 * half + r;
        s1[node] = p1[r];
        s2[node] = p2[r];
      }
    }
  }
}

// ---------------------------------------------------------------------------
// Kernel 2: fully fused per-edge pipeline, 16 edges per wave
// ---------------------------------------------------------------------------
__global__ void edge_kernel(const float* __restrict__ nodeF,
                            const float* __restrict__ edgeF,
                            const float* __restrict__ We,
                            const float* __restrict__ Web,
                            const float* __restrict__ Wm,
                            const int* __restrict__ eidx,
                            const float* __restrict__ s1,
                            const float* __restrict__ s2,
                            float* __restrict__ out) {
  extern __shared__ char smem[];
  _Float16* sWe  = (_Float16*)smem;        // We  B-frags: [n*DIM + k]
  _Float16* sWmT = sWe + DIM * DIM;        // Wm^T A-frags: [m*DIM + k] = Wm[k][m]
  _Float16* sStg = sWmT + DIM * DIM;       // per-wave 16xDIM staging of e_g_l^T
  float* sBias = (float*)(sStg + 8 * 16 * DIM);

  const int tid = threadIdx.x;
  for (int t = tid; t < 2048; t += 256) {
    int n = t >> 4, k0 = (t & 15) * 8;
    v8h p, q;
#pragma unroll
    for (int j = 0; j < 8; ++j) {
      p[j] = (_Float16)We[(k0 + j) * DIM + n];
      q[j] = (_Float16)Wm[(k0 + j) * DIM + n];
    }
    *(v8h*)(sWe  + n * DIM + k0) = p;
    *(v8h*)(sWmT + n * DIM + k0) = q;
  }
  for (int t = tid; t < DIM; t += 256) sBias[t] = Web[t];
  __syncthreads();

  const int lane = tid & 31;
  const int col  = lane & 15;
  const int half = lane >> 4;
  const int wave = tid >> 5;
  _Float16* stg = sStg + wave * 16 * DIM;
  const int nwaves = gridDim.x * 8;

  for (int tile = blockIdx.x * 8 + wave; tile < N_EDGES / 16; tile += nwaves) {
    int zf = opaque_zero();   // per-iteration opaque 0: keep LDS reads in-loop
    int e = tile * 16 + col;
    int i0 = eidx[2 * e + 0];   // source / neighbor node
    int i1 = eidx[2 * e + 1];   // aggregation target
    float att_e = s1[i1] + s2[i0];

    // ---- GEMM 1: upd = edge_tile(16x128) @ We ----
    const float* rp = edgeF + (long)e * DIM;
    v16h afrag[4];
#pragma unroll
    for (int kt = 0; kt < 4; ++kt)
      afrag[kt] = cat16(cvt8(rp + 32 * kt + 8 * half),
                        cvt8(rp + 32 * kt + 16 + 8 * half));

    v8f acc[8];
#pragma unroll
    for (int nt = 0; nt < 8; ++nt) {
      v8f a = {};
      const _Float16* bp = sWe + (16 * nt + col) * DIM + 8 * half + zf;
#pragma unroll
      for (int kt = 0; kt < 4; ++kt) {
        v16h bfrag = cat16(lds8(bp + 32 * kt), lds8(bp + 32 * kt + 16));
        a = wmma_f16(afrag[kt], bfrag, a);
      }
      acc[nt] = a;
    }

    // att scalar for the row each C-register corresponds to (row m = r + 8*half)
    float attm[8];
#pragma unroll
    for (int r = 0; r < 8; ++r) attm[r] = __shfl(att_e, 8 * half + r, 32);

    // ---- softmax over the 128 columns of each edge row ----
    float rmax[8], rsum[8];
#pragma unroll
    for (int r = 0; r < 8; ++r) rmax[r] = -3.4e38f;
#pragma unroll
    for (int nt = 0; nt < 8; ++nt) {
      float bb = sBias[16 * nt + col];
#pragma unroll
      for (int r = 0; r < 8; ++r) {
        float v = attm[r] * (acc[nt][r] + bb);
        acc[nt][r] = v;
        rmax[r] = fmaxf(rmax[r], v);
      }
    }
#pragma unroll
    for (int r = 0; r < 8; ++r) {
#pragma unroll
      for (int m = 1; m < 16; m <<= 1)
        rmax[r] = fmaxf(rmax[r], __shfl_xor(rmax[r], m, 32));
      rsum[r] = 0.f;
    }
#pragma unroll
    for (int nt = 0; nt < 8; ++nt) {
#pragma unroll
      for (int r = 0; r < 8; ++r) {
        float v = __expf(acc[nt][r] - rmax[r]);
        acc[nt][r] = v;
        rsum[r] += v;
      }
    }
#pragma unroll
    for (int r = 0; r < 8; ++r) {
#pragma unroll
      for (int m = 1; m < 16; m <<= 1) rsum[r] += __shfl_xor(rsum[r], m, 32);
      rsum[r] = 1.f / rsum[r];
    }

    // ---- transpose e_g_l through per-wave LDS: stg[edge_local*DIM + feature] ----
#pragma unroll
    for (int nt = 0; nt < 8; ++nt) {
#pragma unroll
      for (int r = 0; r < 8; ++r)
        stg[(8 * half + r) * DIM + 16 * nt + col] =
            (_Float16)(acc[nt][r] * rsum[r]);
    }

    // ---- GEMM 2: z^T = Wm^T (A) @ e_g_l^T (B) -> lane owns one edge ----
    v16h bfragE[4];
#pragma unroll
    for (int kt = 0; kt < 4; ++kt) {
      const _Float16* bp = stg + col * DIM + 32 * kt + 8 * half;
      bfragE[kt] = cat16(lds8(bp), lds8(bp + 16));
    }

    v8f zacc[8];
#pragma unroll
    for (int mt = 0; mt < 8; ++mt) {
      v8f z = {};
      const _Float16* ap = sWmT + (16 * mt + col) * DIM + 8 * half + zf;
#pragma unroll
      for (int kt = 0; kt < 4; ++kt) {
        v16h af = cat16(lds8(ap + 32 * kt), lds8(ap + 32 * kt + 16));
        z = wmma_f16(af, bfragE[kt], z);
      }
      zacc[mt] = z;
    }

    // ---- softmax over 128 features of this lane's edge (split across lane^16) ----
    float m64 = -3.4e38f;
#pragma unroll
    for (int mt = 0; mt < 8; ++mt)
#pragma unroll
      for (int r = 0; r < 8; ++r) m64 = fmaxf(m64, zacc[mt][r]);
    m64 = fmaxf(m64, __shfl_xor(m64, 16, 32));
    float ssum = 0.f;
#pragma unroll
    for (int mt = 0; mt < 8; ++mt) {
#pragma unroll
      for (int r = 0; r < 8; ++r) {
        float v = __expf(zacc[mt][r] - m64);
        zacc[mt][r] = v;
        ssum += v;
      }
    }
    ssum += __shfl_xor(ssum, 16, 32);
    float inv = 1.f / ssum;

    // ---- messages = att2 * nodeF[i0]; scatter-add into out[i1] ----
    const float* nb = nodeF + (long)i0 * DIM;
    float* op = out + (long)i1 * DIM;
#pragma unroll
    for (int mt = 0; mt < 8; ++mt) {
      int f0 = 16 * mt + 8 * half;
      v4f n1 = *(const v4f*)(nb + f0);
      v4f n2 = *(const v4f*)(nb + f0 + 4);
#pragma unroll
      for (int r = 0; r < 4; ++r) {
        atomicAdd(op + f0 + r,     zacc[mt][r]     * inv * n1[r]);
        atomicAdd(op + f0 + 4 + r, zacc[mt][4 + r] * inv * n2[r]);
      }
    }
  }
}

// ---------------------------------------------------------------------------
// Kernel 3: final leaky-ReLU in place
// ---------------------------------------------------------------------------
__global__ void relu_kernel(float* __restrict__ out, int n) {
  int i = blockIdx.x * blockDim.x + threadIdx.x;
  if (i < n) {
    float v = out[i];
    out[i] = v >= 0.f ? v : ALPHA * v;
  }
}

extern "C" void kernel_launch(void* const* d_in, const int* in_sizes, int n_in,
                              void* d_out, int out_size, void* d_ws, size_t ws_size,
                              hipStream_t stream) {
  const float* nodeF = (const float*)d_in[0];
  const float* edgeF = (const float*)d_in[1];
  const float* Wu    = (const float*)d_in[2];
  const float* Wub   = (const float*)d_in[3];
  const float* aw    = (const float*)d_in[4];
  const float* We    = (const float*)d_in[5];
  const float* Web   = (const float*)d_in[6];
  const float* Wm    = (const float*)d_in[7];
  const int*   eidx  = (const int*)d_in[8];
  float* out = (float*)d_out;
  float* s1 = (float*)d_ws;
  float* s2 = s1 + N_NODES;

  size_t smem1 = (size_t)DIM * DIM * sizeof(_Float16) + 3 * DIM * sizeof(float);
  node_score_kernel<<<80, 256, smem1, stream>>>(nodeF, Wu, Wub, aw, s1, s2);

  hipMemsetAsync(d_out, 0, (size_t)N_NODES * DIM * sizeof(float), stream);

  size_t smem3 = (size_t)(2 * DIM * DIM + 8 * 16 * DIM) * sizeof(_Float16) +
                 DIM * sizeof(float);
  edge_kernel<<<640, 256, smem3, stream>>>(nodeF, edgeF, We, Web, Wm, eidx,
                                           s1, s2, out);

  relu_kernel<<<(N_NODES * DIM + 255) / 256, 256, 0, stream>>>(out, N_NODES * DIM);
}
